// LinearScaledDotProductAttention_17695265260071
// MI455X (gfx1250) — compile-verified, matrix operand order checked by reading, not run on hardware
//
#include <hip/hip_runtime.h>
#include <hip/hip_bf16.h>

// ---------------------------------------------------------------------------
// Problem: b=4, h=8, n=4096, e=64, d_model=512
// Phase 1: linear-attention scan -> attn[b,n,512] (stored f16 in ws)
// Phase 2: out = attn @ W^T + bias via v_wmma_f32_16x16x32_f16 with
//          double-buffered global_load_async_to_lds_b128 tile staging.
// ---------------------------------------------------------------------------

typedef __attribute__((ext_vector_type(16))) _Float16 v16h;
typedef __attribute__((ext_vector_type(8)))  float    v8f;

#define B_      4
#define H_      8
#define N_      4096
#define E_      64
#define DM_     512            // h*e
#define BH_     (B_ * H_)      // 32
#define SEG_    64             // segments per (b,h)
#define SLEN_   (N_ / SEG_)    // 64 tokens per segment
#define EPS_    1e-5f

// GEMM tiling
#define GM_     16384          // b*n rows
#define BM_     128
#define BN_     128
#define BK_     32
#define KSTEPS_ (DM_ / BK_)    // 16
#define LDT_    40             // padded LDS row stride (halves) to spread banks

__device__ __forceinline__ float phi_f(float x) {
    // elu(x)+1 : x>0 -> x+1 ; else exp(x)-1+1 = exp(x)
    return x > 0.f ? x + 1.f : __expf(x);
}

// ---------------------------------------------------------------------------
// K1: per-(b,h,seg) partial sums of phi(k) and v over the segment (per channel)
// grid: BH_*SEG_ blocks of 32 threads (1 wave). Lane owns channels 2L, 2L+1.
// ---------------------------------------------------------------------------
__global__ void k1_segsum(const float* __restrict__ k, const float* __restrict__ v,
                          float* __restrict__ ksum, float* __restrict__ vsum) {
    const int blk  = blockIdx.x;
    const int bh   = blk / SEG_;
    const int seg  = blk % SEG_;
    const int lane = threadIdx.x;
    const int c0   = lane * 2;

    float ks0 = 0.f, ks1 = 0.f, vs0 = 0.f, vs1 = 0.f;
    const size_t base = ((size_t)bh * N_ + (size_t)seg * SLEN_) * E_ + c0;
    for (int i = 0; i < SLEN_; ++i) {
        const float2 kk = *(const float2*)(k + base + (size_t)i * E_);
        const float2 vv = *(const float2*)(v + base + (size_t)i * E_);
        ks0 += phi_f(kk.x); ks1 += phi_f(kk.y);
        vs0 += vv.x;        vs1 += vv.y;
    }
    const size_t o = ((size_t)bh * SEG_ + seg) * E_ + c0;
    *(float2*)(ksum + o) = make_float2(ks0, ks1);
    *(float2*)(vsum + o) = make_float2(vs0, vs1);
}

// ---------------------------------------------------------------------------
// K2: exclusive prefix over the SEG_ segment sums, in place.
// grid: BH_ blocks of 128 threads; t<64 -> ksum channel t, t>=64 -> vsum.
// ---------------------------------------------------------------------------
__global__ void k2_prefix(float* __restrict__ ksum, float* __restrict__ vsum) {
    const int bh = blockIdx.x;
    const int t  = threadIdx.x;
    float* arr = (t < E_) ? ksum : vsum;
    const int ch = (t < E_) ? t : (t - E_);
    float run = 0.f;
    for (int seg = 0; seg < SEG_; ++seg) {
        const size_t idx = ((size_t)bh * SEG_ + seg) * E_ + ch;
        const float val = arr[idx];
        arr[idx] = run;
        run += val;
    }
}

// ---------------------------------------------------------------------------
// K3: rescan each segment with its exclusive prefix; emit f16 activations in
// [b, n, h*64] layout. s = sum_e qf*kc via wave32 shfl reduction.
// ---------------------------------------------------------------------------
__global__ void k3_scan(const float* __restrict__ q, const float* __restrict__ k,
                        const float* __restrict__ v,
                        const float* __restrict__ ksum, const float* __restrict__ vsum,
                        _Float16* __restrict__ act) {
    const int blk  = blockIdx.x;
    const int bh   = blk / SEG_;
    const int seg  = blk % SEG_;
    const int lane = threadIdx.x;
    const int c0   = lane * 2;
    const int b    = bh >> 3;
    const int h    = bh & 7;

    const size_t po = ((size_t)bh * SEG_ + seg) * E_ + c0;
    float2 kc = *(const float2*)(ksum + po);
    float2 vc = *(const float2*)(vsum + po);

    const size_t base = ((size_t)bh * N_ + (size_t)seg * SLEN_) * E_ + c0;
    for (int i = 0; i < SLEN_; ++i) {
        const size_t off = base + (size_t)i * E_;
        const float2 kk = *(const float2*)(k + off);
        const float2 vv = *(const float2*)(v + off);
        const float2 qq = *(const float2*)(q + off);
        kc.x += phi_f(kk.x); kc.y += phi_f(kk.y);
        vc.x += vv.x;        vc.y += vv.y;
        float s = phi_f(qq.x) * kc.x + phi_f(qq.y) * kc.y;
        #pragma unroll
        for (int o = 16; o > 0; o >>= 1) s += __shfl_xor(s, o, 32);
        const float ratio = s / (s + EPS_);
        union { _Float16 hh[2]; unsigned u; } pk;
        pk.hh[0] = (_Float16)(ratio * vc.x);
        pk.hh[1] = (_Float16)(ratio * vc.y);
        const int n = seg * SLEN_ + i;
        const size_t dst = ((size_t)b * N_ + n) * DM_ + h * E_ + c0;
        *(unsigned*)(act + dst) = pk.u;
    }
}

// ---------------------------------------------------------------------------
// K2b: W_fc f32 -> f16 (row-major [out=512][in=512], used as WMMA B^T operand)
// ---------------------------------------------------------------------------
__global__ void k_wcvt(const float* __restrict__ w, _Float16* __restrict__ wh) {
    const int i = blockIdx.x * blockDim.x + threadIdx.x;   // 0 .. 262143
    wh[i] = (_Float16)w[i];
}

// ---------------------------------------------------------------------------
// CDNA5 async global->LDS copy (16B per lane), tracked by ASYNCcnt.
// GVS mode: vdst = LDS byte address (VGPR), vaddr = u32 byte offset (VGPR),
// saddr = 64-bit base (SGPR pair).
// ---------------------------------------------------------------------------
__device__ __forceinline__ void async_copy16(unsigned lds_addr, const void* base,
                                             unsigned byte_off) {
    asm volatile("global_load_async_to_lds_b128 %0, %1, %2 offset:0"
                 :: "v"(lds_addr), "v"(byte_off), "s"(base)
                 : "memory");
}
__device__ __forceinline__ void wait_async0() {
    asm volatile("s_wait_asynccnt 0" ::: "memory");
}

// ---------------------------------------------------------------------------
// K4: GEMM  C[16384,512] = A_f16[16384,512] * W[512,512]^T + bias
// block = 256 threads (8 waves); block tile 128x128; wave tile 32x64
// -> 8 x v_wmma_f32_16x16x32_f16 per K-step, 16 K-steps, double-buffered
//    async tile staging.
// ---------------------------------------------------------------------------
__device__ __forceinline__ v16h load_frag(const _Float16* p) {
    // fragment = two contiguous 16B chunks 16 halves apart (K 0-7 / 16-23 style)
    union { v16h v; _Float16 e[16]; } fr;
    *(uint4*)&fr.e[0] = *(const uint4*)(p);
    *(uint4*)&fr.e[8] = *(const uint4*)(p + 16);
    return fr.v;
}

__global__ void k4_gemm(const _Float16* __restrict__ A, const _Float16* __restrict__ Wh,
                        const float* __restrict__ bias, float* __restrict__ out) {
    __shared__ _Float16 lsA[2][BM_ * LDT_];
    __shared__ _Float16 lsB[2][BN_ * LDT_];

    const int tid   = threadIdx.x;
    const int mBase = blockIdx.y * BM_;
    const int nBase = blockIdx.x * BN_;

    const int lane = tid & 31;
    const int warp = tid >> 5;          // 0..7
    const int wm   = warp & 3;          // 4 waves along M -> 32 rows each
    const int wn   = warp >> 2;         // 2 waves along N -> 64 cols each
    const int half = lane >> 4;
    const int l16  = lane & 15;

    // two 16B chunks per thread per tile: chunk ids tid and tid+256
    const int row0 = tid >> 2;                 // 0..63
    const int cc0  = (tid & 3) * 8;            // halves within 32-wide K slab
    const int row1 = row0 + 64;                // 64..127

    v8f acc[2][4] = {};

    // ---- stage K-step kb into LDS buffer `buf` (async, per-lane 16B) -------
    auto stage = [&](int kb, int buf) {
        const unsigned ka = (unsigned)(((size_t)(mBase + row0) * DM_ + kb * BK_ + cc0) * 2);
        const unsigned kb_ = (unsigned)(((size_t)(mBase + row1) * DM_ + kb * BK_ + cc0) * 2);
        const unsigned wa = (unsigned)(((size_t)(nBase + row0) * DM_ + kb * BK_ + cc0) * 2);
        const unsigned wb = (unsigned)(((size_t)(nBase + row1) * DM_ + kb * BK_ + cc0) * 2);
        async_copy16((unsigned)(uintptr_t)&lsA[buf][row0 * LDT_ + cc0], A, ka);
        async_copy16((unsigned)(uintptr_t)&lsA[buf][row1 * LDT_ + cc0], A, kb_);
        async_copy16((unsigned)(uintptr_t)&lsB[buf][row0 * LDT_ + cc0], Wh, wa);
        async_copy16((unsigned)(uintptr_t)&lsB[buf][row1 * LDT_ + cc0], Wh, wb);
    };

    stage(0, 0);
    wait_async0();
    __syncthreads();

    for (int kb = 0; kb < KSTEPS_; ++kb) {
        const int cur = kb & 1;
        if (kb + 1 < KSTEPS_) stage(kb + 1, cur ^ 1);

        v16h af[2], bf[4];
        #pragma unroll
        for (int t = 0; t < 2; ++t)
            af[t] = load_frag(&lsA[cur][(wm * 32 + t * 16 + l16) * LDT_ + half * 8]);
        #pragma unroll
        for (int u = 0; u < 4; ++u)
            bf[u] = load_frag(&lsB[cur][(wn * 64 + u * 16 + l16) * LDT_ + half * 8]);

        #pragma unroll
        for (int t = 0; t < 2; ++t)
            #pragma unroll
            for (int u = 0; u < 4; ++u)
                acc[t][u] = __builtin_amdgcn_wmma_f32_16x16x32_f16(
                    false, af[t], false, bf[u], (short)0, acc[t][u], false, false);

        wait_async0();       // own async copies into next buffer have landed
        __syncthreads();     // everyone's have landed
    }

    // --- epilogue: D layout lane L -> row rr + 8*(L/16), col L%16
    #pragma unroll
    for (int t = 0; t < 2; ++t) {
        #pragma unroll
        for (int u = 0; u < 4; ++u) {
            const int col = nBase + wn * 64 + u * 16 + l16;
            const float bv = bias[col];
            const int rowBase = mBase + wm * 32 + t * 16 + 8 * half;
            #pragma unroll
            for (int rr = 0; rr < 8; ++rr)
                out[(size_t)(rowBase + rr) * DM_ + col] = acc[t][u][rr] + bv;
        }
    }
}

// ---------------------------------------------------------------------------
extern "C" void kernel_launch(void* const* d_in, const int* in_sizes, int n_in,
                              void* d_out, int out_size, void* d_ws, size_t ws_size,
                              hipStream_t stream) {
    const float* q   = (const float*)d_in[0];
    const float* k   = (const float*)d_in[1];
    const float* v   = (const float*)d_in[2];
    // d_in[3] = mask (unused)
    const float* Wfc = (const float*)d_in[4];
    const float* bfc = (const float*)d_in[5];
    float* out = (float*)d_out;

    char* wsb = (char*)d_ws;
    _Float16* act  = (_Float16*)(wsb);                                   // 16 MB
    _Float16* wh   = (_Float16*)(wsb + (size_t)GM_ * DM_ * 2);           // 0.5 MB
    float*    ksum = (float*)(wsb + (size_t)GM_ * DM_ * 2 + (size_t)DM_ * DM_ * 2);
    float*    vsum = ksum + (size_t)BH_ * SEG_ * E_;                     // 0.5 MB each

    k1_segsum<<<dim3(BH_ * SEG_), dim3(32), 0, stream>>>(k, v, ksum, vsum);
    k2_prefix<<<dim3(BH_), dim3(128), 0, stream>>>(ksum, vsum);
    k_wcvt   <<<dim3((DM_ * DM_) / 256), dim3(256), 0, stream>>>(Wfc, wh);
    k3_scan  <<<dim3(BH_ * SEG_), dim3(32), 0, stream>>>(q, k, v, ksum, vsum, act);
    k4_gemm  <<<dim3(DM_ / BN_, GM_ / BM_), dim3(256), 0, stream>>>(act, wh, bfc, out);
}
